// GNN_47029891891685
// MI455X (gfx1250) — matrix operand order repaired and checked by reading, not compiled, not run on hardware
//
#include <hip/hip_runtime.h>

typedef __attribute__((ext_vector_type(16))) _Float16 v16h;
typedef __attribute__((ext_vector_type(8)))  float    v8f;

#define NEG_SLOPE 0.2f
#define BN_EPS    1e-5f

__device__ __forceinline__ float lrelu(float t) { return t > 0.f ? t : NEG_SLOPE * t; }

// Order-preserving float <-> uint encoding so segment-max can use native u32 atomicMax.
// Any finite float encodes to > 0, so zero-initialized buffers act as -inf.
__device__ __forceinline__ unsigned fenc(float f) {
  unsigned u = __float_as_uint(f);
  return (u & 0x80000000u) ? ~u : (u | 0x80000000u);
}
__device__ __forceinline__ float fdec(unsigned u) {
  u = (u & 0x80000000u) ? (u ^ 0x80000000u) : ~u;
  return __uint_as_float(u);
}

__global__ void fill_zero_kernel(float* __restrict__ p, long long n) {
  long long i = (long long)blockIdx.x * blockDim.x + threadIdx.x;
  if (i < n) p[i] = 0.f;
}

// ---------------- WMMA GEMM: C[N,FOUT] = A[N,K] @ W[K,FOUT] (f16 in, f32 acc) -------------
// One wave per 16x16 output tile; whole W staged in LDS per 256-thread block.
// FUSE: 0 = none, 1 = +bias, 2 = +bias then ReLU.
template<int K, int FOUT, int FUSE>
__global__ __launch_bounds__(256)
void wmma_gemm_kernel(const float* __restrict__ A, const float* __restrict__ W,
                      const float* __restrict__ bias, float* __restrict__ C, int N) {
  constexpr int LDSW = FOUT + 4;                 // pad to spread LDS banks across k-rows
  __shared__ float lW[K * LDSW];

  // ---- cooperative stage of W (K*FOUT floats) into LDS ----
  for (int i = threadIdx.x; i < K * FOUT; i += 256) {
    int kk = i / FOUT, cc = i - kk * FOUT;
    lW[kk * LDSW + cc] = W[i];
  }
  __syncthreads();

  int wave   = (int)((blockIdx.x * (long long)blockDim.x + threadIdx.x) >> 5);
  int lane   = threadIdx.x & 31;
  constexpr int tilesN = FOUT >> 4;
  int tileM  = wave / tilesN;
  int tileN  = wave - tileM * tilesN;
  int row0   = tileM << 4;
  if (row0 >= N) return;                         // wave-uniform guard (after barrier)
  int col0 = tileN << 4;
  int half = lane >> 4;                          // 0: lanes 0-15, 1: lanes 16-31
  int lm   = lane & 15;
  int am   = row0 + lm; if (am >= N) am = N - 1; // clamp A-row reads
  const float* arow = A + (size_t)am * K;
  int bcol = col0 + lm;

  v8f acc = {};
#pragma unroll
  for (int k0 = 0; k0 < K; k0 += 32) {
    // A fragment 16x32 f16: lane lm (half h): halves 0..7  -> K = k0 + 8h + i
    //                                         halves 8..15 -> K = k0 + 16 + 8h + (i-8)
    const float4* ap0 = (const float4*)(arow + k0 + half * 8);
    const float4* ap1 = (const float4*)(arow + k0 + half * 8 + 16);
    float4 q0 = ap0[0], q1 = ap0[1], q2 = ap1[0], q3 = ap1[1];
    v16h af, bf;
    af[0]  = (_Float16)q0.x; af[1]  = (_Float16)q0.y; af[2]  = (_Float16)q0.z; af[3]  = (_Float16)q0.w;
    af[4]  = (_Float16)q1.x; af[5]  = (_Float16)q1.y; af[6]  = (_Float16)q1.z; af[7]  = (_Float16)q1.w;
    af[8]  = (_Float16)q2.x; af[9]  = (_Float16)q2.y; af[10] = (_Float16)q2.z; af[11] = (_Float16)q2.w;
    af[12] = (_Float16)q3.x; af[13] = (_Float16)q3.y; af[14] = (_Float16)q3.z; af[15] = (_Float16)q3.w;
    // B fragment 32x16 f16 from LDS: lane holds column bcol; same K order as A fragment
    const float* bp = lW + (k0 + half * 8) * LDSW + bcol;
#pragma unroll
    for (int i = 0; i < 16; ++i)
      bf[i] = (_Float16)bp[(i + (i & 8)) * LDSW];
    acc = __builtin_amdgcn_wmma_f32_16x16x32_f16(false, af, false, bf,
                                                 (short)0, acc, false, false);
  }

  // ---- epilogue: C/D layout: vgpr v, lane: row = row0 + v + 8*half, col = bcol ----
  float bval = 0.f;
  if constexpr (FUSE != 0) bval = bias[bcol];
  float* crow = C + (size_t)row0 * FOUT + bcol;
  if (row0 + 16 <= N) {                           // full tile: unguarded stores
#pragma unroll
    for (int v = 0; v < 8; ++v) {
      float o = acc[v];
      if constexpr (FUSE != 0) o += bval;
      if constexpr (FUSE == 2) o = fmaxf(o, 0.f);
      crow[(size_t)(v + half * 8) * FOUT] = o;
    }
  } else {
#pragma unroll
    for (int v = 0; v < 8; ++v) {
      int r = row0 + v + half * 8;
      if (r < N) {
        float o = acc[v];
        if constexpr (FUSE != 0) o += bval;
        if constexpr (FUSE == 2) o = fmaxf(o, 0.f);
        C[(size_t)r * FOUT + bcol] = o;
      }
    }
  }
}

// ------------- per-(node,head) attention logit components: a = <xh[n,h,:], att[h,:]> ------
__global__ void attn_scores_kernel(const float* __restrict__ xh,
                                   const float* __restrict__ atts,
                                   const float* __restrict__ attd,
                                   float* __restrict__ a_s, float* __restrict__ a_d,
                                   int N, int F, int C) {
  int t = blockIdx.x * blockDim.x + threadIdx.x;
  if (t >= N * 4) return;
  int n = t >> 2, h = t & 3;
  const float* xr = xh + (size_t)n * F + h * C;
  const float* vs = atts + h * C;
  const float* vd = attd + h * C;
  float ss = 0.f, dd = 0.f;
  for (int c = 0; c < C; ++c) { float v = xr[c]; ss += v * vs[c]; dd += v * vd[c]; }
  a_s[t] = ss; a_d[t] = dd;
}

// ------------- edge pass 1: segment max of leaky_relu(a_src[s]+a_dst[d]) per head ---------
__global__ void edge_max_kernel(const int* __restrict__ ei, int E, int N,
                                const float* __restrict__ a_s, const float* __restrict__ a_d,
                                unsigned* __restrict__ menc) {
  int e = blockIdx.x * blockDim.x + threadIdx.x;
  int Et = E + N;
  if (e >= Et) return;
  int s, d;
  if (e < E) { s = ei[e]; d = ei[E + e]; } else { s = d = e - E; }
  float4 vs = *(const float4*)(a_s + (size_t)s * 4);
  float4 vd = *(const float4*)(a_d + (size_t)d * 4);
  unsigned* md = menc + (size_t)d * 4;
  atomicMax(md + 0, fenc(lrelu(vs.x + vd.x)));
  atomicMax(md + 1, fenc(lrelu(vs.y + vd.y)));
  atomicMax(md + 2, fenc(lrelu(vs.z + vd.z)));
  atomicMax(md + 3, fenc(lrelu(vs.w + vd.w)));
}

// ------------- edge pass 2: segment sum of exp(e - m[dst]) -------------------------------
__global__ void edge_sum_kernel(const int* __restrict__ ei, int E, int N,
                                const float* __restrict__ a_s, const float* __restrict__ a_d,
                                const unsigned* __restrict__ menc, float* __restrict__ ssum) {
  int e = blockIdx.x * blockDim.x + threadIdx.x;
  int Et = E + N;
  if (e >= Et) return;
  int s, d;
  if (e < E) { s = ei[e]; d = ei[E + e]; } else { s = d = e - E; }
  float4 vs = *(const float4*)(a_s + (size_t)s * 4);
  float4 vd = *(const float4*)(a_d + (size_t)d * 4);
  const unsigned* md = menc + (size_t)d * 4;
  float* sd = ssum + (size_t)d * 4;
  atomicAdd(sd + 0, __expf(lrelu(vs.x + vd.x) - fdec(md[0])));
  atomicAdd(sd + 1, __expf(lrelu(vs.y + vd.y) - fdec(md[1])));
  atomicAdd(sd + 2, __expf(lrelu(vs.z + vd.z) - fdec(md[2])));
  atomicAdd(sd + 3, __expf(lrelu(vs.w + vd.w) - fdec(md[3])));
}

// ------------- edge pass 3: out[dst] += alpha * xh[src]  (thread per edge,head) ----------
__global__ void edge_aggr_kernel(const int* __restrict__ ei, int E, int N,
                                 const float* __restrict__ a_s, const float* __restrict__ a_d,
                                 const unsigned* __restrict__ menc,
                                 const float* __restrict__ ssum,
                                 const float* __restrict__ xh, float* __restrict__ agg,
                                 int F, int C) {
  int t = blockIdx.x * blockDim.x + threadIdx.x;
  int Et = E + N;
  int e = t >> 2, h = t & 3;
  if (e >= Et) return;
  int s, d;
  if (e < E) { s = ei[e]; d = ei[E + e]; } else { s = d = e - E; }
  float a  = lrelu(a_s[(size_t)s * 4 + h] + a_d[(size_t)d * 4 + h]);
  float ee = __expf(a - fdec(menc[(size_t)d * 4 + h]));
  float alpha = ee / (ssum[(size_t)d * 4 + h] + 1e-16f);
  const float* xr = xh  + (size_t)s * F + h * C;
  float*       od = agg + (size_t)d * F + h * C;
  for (int c = 0; c < C; c += 4) {
    float4 v = *(const float4*)(xr + c);
    atomicAdd(od + c + 0, alpha * v.x);
    atomicAdd(od + c + 1, alpha * v.y);
    atomicAdd(od + c + 2, alpha * v.z);
    atomicAdd(od + c + 3, alpha * v.w);
  }
}

// ------------- bias + BN(eval) + ReLU, in place ------------------------------------------
__global__ void bias_bn_relu_kernel(float* __restrict__ hio, const float* __restrict__ bias,
                                    const float* __restrict__ g, const float* __restrict__ b,
                                    const float* __restrict__ rm, const float* __restrict__ rv,
                                    long long total, int F) {
  long long i = (long long)blockIdx.x * blockDim.x + threadIdx.x;
  if (i >= total) return;
  int f = (int)(i % F);
  float v = hio[i] + bias[f];
  v = (v - rm[f]) * __frsqrt_rn(rv[f] + BN_EPS) * g[f] + b[f];
  hio[i] = v > 0.f ? v : 0.f;
}

// ------------- pooling gate: gate[n] = <h2[n], gW> + gb ; segment max over graphs --------
__global__ void gate_kernel(const float* __restrict__ h2, const float* __restrict__ gW,
                            const float* __restrict__ gb, const int* __restrict__ batch,
                            float* __restrict__ gate, unsigned* __restrict__ gmenc, int N) {
  int n = blockIdx.x * blockDim.x + threadIdx.x;
  if (n >= N) return;
  const float* hr = h2 + (size_t)n * 128;
  float gsum = gb[0];
  for (int k = 0; k < 128; k += 4) {
    float4 v = *(const float4*)(hr + k);
    gsum += v.x * gW[k] + v.y * gW[k + 1] + v.z * gW[k + 2] + v.w * gW[k + 3];
  }
  gate[n] = gsum;
  atomicMax(gmenc + batch[n], fenc(gsum));
}

__global__ void gate_exp_kernel(float* __restrict__ gate, const unsigned* __restrict__ gmenc,
                                float* __restrict__ gs, const int* __restrict__ batch, int N) {
  int n = blockIdx.x * blockDim.x + threadIdx.x;
  if (n >= N) return;
  int b = batch[n];
  float ge = __expf(gate[n] - fdec(gmenc[b]));
  gate[n] = ge;
  atomicAdd(gs + b, ge);
}

__global__ void pool_kernel(const float* __restrict__ ge, const float* __restrict__ gs,
                            const int* __restrict__ batch, const float* __restrict__ h2,
                            float* __restrict__ pooled, int N) {
  long long i = (long long)blockIdx.x * blockDim.x + threadIdx.x;
  if (i >= (long long)N * 128) return;
  int n = (int)(i >> 7), f = (int)(i & 127);
  int b = batch[n];
  float w = ge[n] / (gs[b] + 1e-16f);
  atomicAdd(pooled + (size_t)b * 128 + f, w * h2[(size_t)n * 128 + f]);
}

// ------------- dueling head combine: out = value + adv - mean(adv) -----------------------
__global__ void final_kernel(const float* __restrict__ z, const float* __restrict__ vW,
                             const float* __restrict__ vb, const float* __restrict__ adv,
                             float* __restrict__ out, int G) {
  int i = blockIdx.x * blockDim.x + threadIdx.x;
  if (i >= G * 16) return;
  int g = i >> 4;
  const float* zr = z + (size_t)g * 32;
  float val = vb[0];
  for (int k = 0; k < 32; ++k) val += zr[k] * vW[k];
  const float* ar = adv + (size_t)g * 16;
  float mean = 0.f;
  for (int j = 0; j < 16; ++j) mean += ar[j];
  mean *= (1.f / 16.f);
  out[i] = val + ar[i & 15] - mean;
}

#define CDIV(a, b) (((a) + (b) - 1) / (b))

extern "C" void kernel_launch(void* const* d_in, const int* in_sizes, int n_in,
                              void* d_out, int out_size, void* d_ws, size_t ws_size,
                              hipStream_t stream) {
  const int N = in_sizes[0] / 32;      // 50000
  const int E = in_sizes[1] / 2;       // 1600000
  const int G = out_size / 16;         // 500
  const int Et = E + N;

  const float* x      = (const float*)d_in[0];
  const int*   ei     = (const int*)  d_in[1];
  const int*   batch  = (const int*)  d_in[2];
  const float* W1     = (const float*)d_in[3];
  const float* as1    = (const float*)d_in[4];
  const float* ad1    = (const float*)d_in[5];
  const float* b1     = (const float*)d_in[6];
  const float* bn1g   = (const float*)d_in[7];
  const float* bn1b   = (const float*)d_in[8];
  const float* bn1rm  = (const float*)d_in[9];
  const float* bn1rv  = (const float*)d_in[10];
  const float* W2     = (const float*)d_in[11];
  const float* as2    = (const float*)d_in[12];
  const float* ad2    = (const float*)d_in[13];
  const float* b2     = (const float*)d_in[14];
  const float* bn2g   = (const float*)d_in[15];
  const float* bn2b   = (const float*)d_in[16];
  const float* bn2rm  = (const float*)d_in[17];
  const float* bn2rv  = (const float*)d_in[18];
  const float* gateW  = (const float*)d_in[19];
  const float* gateB  = (const float*)d_in[20];
  const float* fc1W   = (const float*)d_in[21];
  const float* fc1B   = (const float*)d_in[22];
  const float* valW   = (const float*)d_in[23];
  const float* valB   = (const float*)d_in[24];
  const float* advW   = (const float*)d_in[25];
  const float* advB   = (const float*)d_in[26];
  float* out = (float*)d_out;

  // ---- workspace carve-up (floats). Zero-init region is contiguous at the end. ----
  float* p = (float*)d_ws;
  float* xh1  = p; p += (size_t)N * 64;
  float* xh2  = p; p += (size_t)N * 128;
  float* a1s  = p; p += (size_t)N * 4;
  float* a1d  = p; p += (size_t)N * 4;
  float* a2s  = p; p += (size_t)N * 4;
  float* a2d  = p; p += (size_t)N * 4;
  float* gate = p; p += N;
  float* z    = p; p += (size_t)G * 32;
  float* adv  = p; p += (size_t)G * 16;
  float* zero0 = p;
  float*    agg1 = p; p += (size_t)N * 64;    // becomes h1 in place
  float*    agg2 = p; p += (size_t)N * 128;   // becomes h2 in place
  unsigned* m1 = (unsigned*)p; p += (size_t)N * 4;
  float*    s1 = p;            p += (size_t)N * 4;
  unsigned* m2 = (unsigned*)p; p += (size_t)N * 4;
  float*    s2 = p;            p += (size_t)N * 4;
  unsigned* gm = (unsigned*)p; p += G;
  float*    gs = p;            p += G;
  float*    pooled = p;        p += (size_t)G * 128;
  long long nzero = (long long)(p - zero0);

  const int T = 256;
  fill_zero_kernel<<<CDIV(nzero, (long long)T), T, 0, stream>>>(zero0, nzero);

  // ---- conv1: xh1 = x @ W1 (WMMA), scores, edge softmax-aggregate, bias+BN+ReLU ----
  {
    long long waves = (long long)CDIV(N, 16) * (64 / 16);
    wmma_gemm_kernel<32, 64, 0><<<CDIV(waves * 32, (long long)T), T, 0, stream>>>(x, W1, nullptr, xh1, N);
  }
  attn_scores_kernel<<<CDIV(N * 4, T), T, 0, stream>>>(xh1, as1, ad1, a1s, a1d, N, 64, 16);
  edge_max_kernel <<<CDIV(Et, T), T, 0, stream>>>(ei, E, N, a1s, a1d, m1);
  edge_sum_kernel <<<CDIV(Et, T), T, 0, stream>>>(ei, E, N, a1s, a1d, m1, s1);
  edge_aggr_kernel<<<CDIV(Et * 4, T), T, 0, stream>>>(ei, E, N, a1s, a1d, m1, s1, xh1, agg1, 64, 16);
  bias_bn_relu_kernel<<<CDIV((long long)N * 64, (long long)T), T, 0, stream>>>(
      agg1, b1, bn1g, bn1b, bn1rm, bn1rv, (long long)N * 64, 64);

  // ---- conv2: xh2 = h1 @ W2 (WMMA), scores, edge softmax-aggregate, bias+BN+ReLU ----
  {
    long long waves = (long long)CDIV(N, 16) * (128 / 16);
    wmma_gemm_kernel<64, 128, 0><<<CDIV(waves * 32, (long long)T), T, 0, stream>>>(agg1, W2, nullptr, xh2, N);
  }
  attn_scores_kernel<<<CDIV(N * 4, T), T, 0, stream>>>(xh2, as2, ad2, a2s, a2d, N, 128, 32);
  edge_max_kernel <<<CDIV(Et, T), T, 0, stream>>>(ei, E, N, a2s, a2d, m2);
  edge_sum_kernel <<<CDIV(Et, T), T, 0, stream>>>(ei, E, N, a2s, a2d, m2, s2);
  edge_aggr_kernel<<<CDIV(Et * 4, T), T, 0, stream>>>(ei, E, N, a2s, a2d, m2, s2, xh2, agg2, 128, 32);
  bias_bn_relu_kernel<<<CDIV((long long)N * 128, (long long)T), T, 0, stream>>>(
      agg2, b2, bn2g, bn2b, bn2rm, bn2rv, (long long)N * 128, 128);

  // ---- attentional pooling over graphs ----
  gate_kernel    <<<CDIV(N, T), T, 0, stream>>>(agg2, gateW, gateB, batch, gate, gm, N);
  gate_exp_kernel<<<CDIV(N, T), T, 0, stream>>>(gate, gm, gs, batch, N);
  pool_kernel    <<<CDIV((long long)N * 128, (long long)T), T, 0, stream>>>(gate, gs, batch, agg2, pooled, N);

  // ---- heads: z = relu(pooled@fc1 + b) [WMMA], adv = z@advW + b [WMMA], dueling combine ----
  {
    long long waves = (long long)CDIV(G, 16) * (32 / 16);
    wmma_gemm_kernel<128, 32, 2><<<CDIV(waves * 32, (long long)T), T, 0, stream>>>(pooled, fc1W, fc1B, z, G);
  }
  {
    long long waves = (long long)CDIV(G, 16) * (16 / 16);
    wmma_gemm_kernel<32, 16, 1><<<CDIV(waves * 32, (long long)T), T, 0, stream>>>(z, advW, advB, adv, G);
  }
  final_kernel<<<CDIV(G * 16, T), T, 0, stream>>>(z, valW, valB, adv, out, G);
}
